// HarmonicStructureDiscriminator_64123861729944
// MI455X (gfx1250) — matrix operand order repaired and checked by reading, not compiled, not run on hardware
//
#include <hip/hip_runtime.h>

typedef __attribute__((ext_vector_type(16))) __bf16 v16bf;
typedef __attribute__((ext_vector_type(8)))  __bf16 v8bf;
typedef __attribute__((ext_vector_type(8)))  float  v8f;
typedef __attribute__((ext_vector_type(4)))  int    i32x4;
typedef __attribute__((ext_vector_type(8)))  int    i32x8;

#define F_DIM 257
#define T_DIM 257
#define BATCH 4

static __device__ inline unsigned short f2bf(float x) {
    union { float f; unsigned u; } cv; cv.f = x;
    unsigned u = cv.u;
    unsigned r = u + 0x7FFFu + ((u >> 16) & 1u);
    return (unsigned short)(r >> 16);
}
static __device__ inline float bf2f(unsigned short h) {
    union { unsigned u; float f; } cv; cv.u = ((unsigned)h) << 16;
    return cv.f;
}

// ---------------------------------------------------------------------------
// 1. Weight-norm + conversion to bf16 GEMM layouts.
//    Layer 0 : v_h (64,14,1,7)  -> Wh[oc][k], k = kw*16 + ic, Kpad = 128
//    Layer 1..8: vs[i] (64,64,3,3) -> Wmid[(i*64+oc)][k], k = (kh*3+kw)*64+ic, Kpad=576
//    Layer 9 : v_last (1,64,3,3) -> Wlast[k], same k mapping
// ---------------------------------------------------------------------------
__global__ __launch_bounds__(64)
void weights_prep_kernel(const float* __restrict__ v_h, const float* __restrict__ g_h,
                         const float* __restrict__ vs,  const float* __restrict__ gs,
                         const float* __restrict__ v_last, const float* __restrict__ g_last,
                         unsigned short* __restrict__ Wh,
                         unsigned short* __restrict__ Wmid,
                         unsigned short* __restrict__ Wlast)
{
    int layer = blockIdx.x >> 6;
    int oc    = blockIdx.x & 63;
    int t     = threadIdx.x;
    if (layer == 9 && oc != 0) return;

    __shared__ float red[64];
    __shared__ float snorm;

    const float* v; int nElems; float g;
    if (layer == 0)      { v = v_h + oc * 98;                       nElems = 98;  g = g_h[oc]; }
    else if (layer <= 8) { int i = layer - 1;
                           v = vs + ((size_t)(i * 64 + oc)) * 576;  nElems = 576; g = gs[i * 64 + oc]; }
    else                 { v = v_last;                              nElems = 576; g = g_last[0]; }

    float acc = 0.f;
    for (int j = t; j < nElems; j += 64) { float x = v[j]; acc += x * x; }
    red[t] = acc;
    __syncthreads();
    if (t == 0) {
        float s = 0.f;
        for (int j = 0; j < 64; ++j) s += red[j];
        snorm = sqrtf(s);
    }
    __syncthreads();
    float scale = g / snorm;

    if (layer == 0) {
        unsigned short* row = Wh + (size_t)oc * 128;
        for (int k = t; k < 128; k += 64) {
            int kw = k >> 4, ic = k & 15;
            float val = 0.f;
            if (kw < 7 && ic < 14) val = scale * v_h[((size_t)oc * 14 + ic) * 7 + kw];
            row[k] = f2bf(val);
        }
    } else if (layer <= 8) {
        int i = layer - 1;
        unsigned short* row = Wmid + ((size_t)(i * 64 + oc)) * 576;
        for (int k = t; k < 576; k += 64) {
            int tap = k >> 6, ic = k & 63;
            int kh = tap / 3, kw = tap - kh * 3;
            float val = scale * vs[(((size_t)(i * 64 + oc) * 64 + ic) * 3 + kh) * 3 + kw];
            row[k] = f2bf(val);
        }
    } else {
        for (int k = t; k < 576; k += 64) {
            int tap = k >> 6, ic = k & 63;
            int kh = tap / 3, kw = tap - kh * 3;
            float val = scale * v_last[((size_t)ic * 3 + kh) * 3 + kw];
            Wlast[k] = f2bf(val);
        }
    }
}

// ---------------------------------------------------------------------------
// 2. STFT: direct windowed rDFT with LDS cosine table (period 512).
// ---------------------------------------------------------------------------
__global__ __launch_bounds__(256)
void stft_kernel(const float* __restrict__ x, float* __restrict__ h)
{
    __shared__ float ctab[512];
    for (int i = threadIdx.x; i < 512; i += 256)
        ctab[i] = cosf(6.283185307179586f * (float)i / 512.0f);
    __syncthreads();

    int gid = blockIdx.x * 256 + threadIdx.x;
    const int total = BATCH * F_DIM * T_DIM;
    if (gid >= total) return;
    int t = gid % T_DIM; int r0 = gid / T_DIM;
    int f = r0 % F_DIM;  int b = r0 / F_DIM;

    const float* xb = x + (size_t)b * 65536;
    float re = 0.f, im = 0.f;
    for (int n = 0; n < 512; ++n) {
        int p = t * 256 + n - 256;
        int s = (p < 0) ? -p : ((p >= 65536) ? (131070 - p) : p);
        float w  = 0.5f - 0.5f * ctab[n];
        float wx = w * xb[s];
        int r = (f * n) & 511;
        re += wx * ctab[r];
        im -= wx * ctab[(r + 384) & 511];
    }
    h[(((size_t)b * 2 + 0) * F_DIM + f) * T_DIM + t] = re;
    h[(((size_t)b * 2 + 1) * F_DIM + f) * T_DIM + t] = im;
}

// ---------------------------------------------------------------------------
// 3. Log-shift lowering: low[b][ch][f][t], ch = k*2 + c (16 channels, 14..15 zero)
// ---------------------------------------------------------------------------
__global__ __launch_bounds__(256)
void lower_kernel(const float* __restrict__ h, unsigned short* __restrict__ low)
{
    int gid = blockIdx.x * 256 + threadIdx.x;
    const int total = BATCH * 16 * F_DIM * T_DIM;
    if (gid >= total) return;
    int t = gid % T_DIM; int r0 = gid / T_DIM;
    int f = r0 % F_DIM;  r0 /= F_DIM;
    int ch = r0 & 15;    int b = r0 >> 4;

    size_t oidx = (((size_t)b * 16 + ch) * F_DIM + f) * T_DIM + t;
    if (ch >= 14) { low[oidx] = 0; return; }
    int k = ch >> 1, c = ch & 1;
    float shift = 1000.0f * logf(7.0f / (float)(k + 1));
    float src = (float)f - shift;
    float lo = floorf(src);
    int li = (int)lo;
    float frac = src - lo;
    float g0 = (li     >= 0 && li     < F_DIM) ? h[(((size_t)b * 2 + c) * F_DIM + li    ) * T_DIM + t] : 0.f;
    float g1 = (li + 1 >= 0 && li + 1 < F_DIM) ? h[(((size_t)b * 2 + c) * F_DIM + li + 1) * T_DIM + t] : 0.f;
    low[oidx] = f2bf((1.f - frac) * g0 + frac * g1);
}

// ---------------------------------------------------------------------------
// 4. Implicit-GEMM conv via v_wmma_f32_16x16x32_bf16.
//    Weights (64 x Kpad bf16) staged to LDS once per block by the Tensor Data
//    Mover (tensor_load_to_lds, 2-D D#).  B tiles (64K x 32N) double-buffered
//    in LDS with one barrier per iteration; 2 WMMAs per wave per iteration.
// ---------------------------------------------------------------------------
__global__ __launch_bounds__(256)
void conv_wmma_kernel(const unsigned short* __restrict__ in,
                      const unsigned short* __restrict__ W,
                      const float* __restrict__ bias,
                      unsigned short* __restrict__ out,
                      int icShift, int KW, int cX, int cY, int dil,
                      int Kpad, float slope)
{
    __shared__ __align__(64)  unsigned short Btile[2][32 * 72];  // [buf][n][k], k-stride 72
    __shared__ __align__(128) unsigned short Wlds[64 * 576];     // max Kpad

    const int nTile = (T_DIM + 31) / 32;   // 9
    int idx = blockIdx.x;
    int tt = idx % nTile; idx /= nTile;
    int f  = idx % F_DIM; idx /= F_DIM;
    int b  = idx;
    int t0 = tt * 32;

    int tid  = threadIdx.x;
    int lane = tid & 31;
    int wave = tid >> 5;
    int ocb  = (wave & 3) * 16;
    int nb   = (wave >> 2) * 16;

    const int Cact   = 1 << icShift;
    const int icMask = Cact - 1;
    const int nK64   = Kpad >> 6;

    __builtin_prefetch(W, 0, 3);   // warm GL2 ahead of the TDM fetch

    // ---- TDM: stage the whole weight matrix into LDS (wave 0 issues it)
    if (wave == 0) {
        unsigned ldsoff = (unsigned)(unsigned long long)(&Wlds[0]);
        unsigned long long ga = (unsigned long long)W;
        i32x4 g0;
        g0[0] = 1;                                                  // count=1, user mode
        g0[1] = (int)ldsoff;                                        // lds_addr
        g0[2] = (int)(unsigned)(ga & 0xFFFFFFFFu);                  // global_addr[31:0]
        g0[3] = (int)(((unsigned)(ga >> 32) & 0x01FFFFFFu)          // global_addr[56:32]
                      | 0x80000000u);                               // type=2
        i32x8 g1;
        unsigned kp = (unsigned)Kpad;
        g1[0] = 0x10000;                       // workgroup_mask=0, data_size=1 (2B)
        g1[1] = (int)((kp & 0xFFFFu) << 16);   // tensor_dim0[15:0]
        g1[2] = (int)((kp >> 16) | (64u << 16));   // tensor_dim0[31:16] | tensor_dim1[15:0]
        g1[3] = (int)((kp & 0xFFFFu) << 16);   // tensor_dim1[31:16]=0 | tile_dim0
        g1[4] = 64;                            // tile_dim1=64, tile_dim2=0
        g1[5] = (int)kp;                       // tensor_dim0_stride[31:0]
        g1[6] = 0;                             // stride0[47:32]=0 | stride1[15:0]=0
        g1[7] = 0;
        asm volatile("tensor_load_to_lds %0, %1" :: "s"(g0), "s"(g1) : "memory");
        __builtin_amdgcn_s_wait_tensorcnt(0);
    }

    // staging decomposition: 256 threads cover 64 K x 32 N (8 N each)
    int sk = tid >> 2;        // 0..63
    int sn = (tid & 3) * 8;   // 0,8,16,24

    // ---- stage B tile 0 (synchronous gather, transposed [n][k] layout)
    {
        int kg  = sk;
        int tap = kg >> icShift;
        int ic  = kg & icMask;
        int kh  = tap / KW;
        int kw  = tap - kh * KW;
        int fi  = f + (kh - cY) * dil;
        bool frowok = (fi >= 0) && (fi < F_DIM);
        const unsigned short* inrow = in + (((size_t)(b * Cact + ic) * F_DIM + fi) * T_DIM);
        int tbase = t0 + sn + (kw - cX) * dil;
        #pragma unroll
        for (int e = 0; e < 8; ++e) {
            int tj = tbase + e;
            unsigned short v = 0;
            if (frowok && tj >= 0 && tj < T_DIM) v = inrow[tj];
            Btile[0][(sn + e) * 72 + sk] = v;
        }
    }
    __syncthreads();   // Wlds (TDM) + Btile[0] visible to everyone

    v8f acc = {};
    for (int j = 0; j < nK64; ++j) {
        // ---- stage next B tile into the other buffer (no barrier yet)
        if (j + 1 < nK64) {
            int kg  = (j + 1) * 64 + sk;
            int tap = kg >> icShift;
            int ic  = kg & icMask;
            int kh  = tap / KW;
            int kw  = tap - kh * KW;
            int fi  = f + (kh - cY) * dil;
            bool frowok = (fi >= 0) && (fi < F_DIM);
            const unsigned short* inrow = in + (((size_t)(b * Cact + ic) * F_DIM + fi) * T_DIM);
            int tbase = t0 + sn + (kw - cX) * dil;
            unsigned short* bt = &Btile[(j + 1) & 1][0];
            #pragma unroll
            for (int e = 0; e < 8; ++e) {
                int tj = tbase + e;
                unsigned short v = 0;
                if (frowok && tj >= 0 && tj < T_DIM) v = inrow[tj];
                bt[(sn + e) * 72 + sk] = v;
            }
        }

        // ---- compute: two 16x16x32 WMMAs on the current buffer
        const unsigned short* bt = &Btile[j & 1][0];
        int hl = lane >> 4;
        int m  = ocb + (lane & 15);
        int n  = nb  + (lane & 15);
        #pragma unroll
        for (int s = 0; s < 2; ++s) {
            int kb = j * 64 + s * 32;
            // A fragment from LDS weights: lane<16 -> K {0..7,16..23}, else {8..15,24..31}
            const unsigned short* wrow = Wlds + (size_t)m * Kpad + kb + hl * 8;
            v8bf alo = *(const v8bf*)(wrow);
            v8bf ahi = *(const v8bf*)(wrow + 16);
            v16bf a;
            #pragma unroll
            for (int i = 0; i < 8; ++i) { a[i] = alo[i]; a[8 + i] = ahi[i]; }
            // B fragment from LDS tile: column n, lane<16 -> K {0..7,16..23}, else {8..15,24..31}
            const unsigned short* brow = bt + n * 72 + s * 32 + hl * 8;
            v8bf blo = *(const v8bf*)(brow);
            v8bf bhi = *(const v8bf*)(brow + 16);
            v16bf bf;
            #pragma unroll
            for (int i = 0; i < 8; ++i) { bf[i] = blo[i]; bf[8 + i] = bhi[i]; }

            acc = __builtin_amdgcn_wmma_f32_16x16x32_bf16(false, a, false, bf,
                                                          (short)0, acc, false, false);
        }
        __syncthreads();   // staged tile visible; current buffer free for reuse
    }

    // ---- epilogue: bias + leaky ReLU, bf16 store.  C/D: M = r + 8*(lane>=16), N = lane%16
    int hl = lane >> 4;
    int t  = t0 + nb + (lane & 15);
    #pragma unroll
    for (int r = 0; r < 8; ++r) {
        int m = ocb + r + hl * 8;
        float v = acc[r] + bias[m];
        if (v < 0.f) v *= slope;
        if (t < T_DIM)
            out[(((size_t)b * 64 + m) * F_DIM + f) * T_DIM + t] = f2bf(v);
    }
}

// ---------------------------------------------------------------------------
// 5. Final 3x3 conv, 64 -> 1 channel, f32 output (no activation).
// ---------------------------------------------------------------------------
__global__ __launch_bounds__(256)
void last_conv_kernel(const unsigned short* __restrict__ act,
                      const unsigned short* __restrict__ Wlast,
                      const float* __restrict__ b_last,
                      float* __restrict__ outp)
{
    int gid = blockIdx.x * 256 + threadIdx.x;
    const int total = BATCH * F_DIM * T_DIM;
    if (gid >= total) return;
    int t = gid % T_DIM; int r0 = gid / T_DIM;
    int f = r0 % F_DIM;  int b = r0 / F_DIM;

    float acc = b_last[0];
    for (int tap = 0; tap < 9; ++tap) {
        int kh = tap / 3, kw = tap - kh * 3;
        int fi = f + kh - 1, tj = t + kw - 1;
        if (fi < 0 || fi >= F_DIM || tj < 0 || tj >= T_DIM) continue;
        const unsigned short* wrow = Wlast + tap * 64;
        for (int ic = 0; ic < 64; ++ic) {
            float av = bf2f(act[(((size_t)b * 64 + ic) * F_DIM + fi) * T_DIM + tj]);
            acc += av * bf2f(wrow[ic]);
        }
    }
    outp[(size_t)b * 66049 + f * 257 + t] = acc;
}

// ---------------------------------------------------------------------------
extern "C" void kernel_launch(void* const* d_in, const int* in_sizes, int n_in,
                              void* d_out, int out_size, void* d_ws, size_t ws_size,
                              hipStream_t stream) {
    const float* x      = (const float*)d_in[0];
    const float* v_h    = (const float*)d_in[1];
    const float* g_h    = (const float*)d_in[2];
    const float* b_h    = (const float*)d_in[3];
    const float* vs     = (const float*)d_in[4];
    const float* gs     = (const float*)d_in[5];
    const float* bs     = (const float*)d_in[6];
    const float* v_last = (const float*)d_in[7];
    const float* g_last = (const float*)d_in[8];
    const float* b_last = (const float*)d_in[9];
    float* outp = (float*)d_out;

    char* ws = (char*)d_ws;
    size_t off = 0;
    auto alloc = [&](size_t bytes) { size_t o = off; off = (off + bytes + 255) & ~(size_t)255; return o; };
    const size_t nPix = (size_t)F_DIM * T_DIM;               // 66049
    float*          h    = (float*)         (ws + alloc(BATCH * 2  * nPix * sizeof(float)));
    unsigned short* low  = (unsigned short*)(ws + alloc(BATCH * 16 * nPix * 2));
    unsigned short* actA = (unsigned short*)(ws + alloc(BATCH * 64 * nPix * 2));
    unsigned short* actB = (unsigned short*)(ws + alloc(BATCH * 64 * nPix * 2));
    unsigned short* Wh   = (unsigned short*)(ws + alloc(64 * 128 * 2));
    unsigned short* Wmid = (unsigned short*)(ws + alloc((size_t)8 * 64 * 576 * 2));
    unsigned short* Wl   = (unsigned short*)(ws + alloc(576 * 2));

    weights_prep_kernel<<<dim3(640), dim3(64), 0, stream>>>(
        v_h, g_h, vs, gs, v_last, g_last, Wh, Wmid, Wl);

    const int totPix = BATCH * F_DIM * T_DIM;                // 264196
    stft_kernel<<<dim3((totPix + 255) / 256), dim3(256), 0, stream>>>(x, h);

    const int totLow = BATCH * 16 * F_DIM * T_DIM;
    lower_kernel<<<dim3((totLow + 255) / 256), dim3(256), 0, stream>>>(h, low);

    const int convGrid = BATCH * F_DIM * ((T_DIM + 31) / 32);   // 9252
    conv_wmma_kernel<<<dim3(convGrid), dim3(256), 0, stream>>>(
        low, Wh, b_h, actA, /*icShift*/4, /*KW*/7, /*cX*/3, /*cY*/0, /*dil*/1, /*Kpad*/128, 0.2f);
    for (int i = 0; i < 8; ++i) {
        const unsigned short* src = (i & 1) ? actB : actA;
        unsigned short*       dst = (i & 1) ? actA : actB;
        conv_wmma_kernel<<<dim3(convGrid), dim3(256), 0, stream>>>(
            src, Wmid + (size_t)i * 64 * 576, bs + i * 64, dst,
            /*icShift*/6, /*KW*/3, /*cX*/1, /*cY*/1, /*dil*/i + 1, /*Kpad*/576, 0.2f);
    }

    last_conv_kernel<<<dim3((totPix + 255) / 256), dim3(256), 0, stream>>>(
        actA, Wl, b_last, outp);
}
